// PointMixer_34153579938133
// MI455X (gfx1250) — compile-verified, matrix operand order checked by reading, not run on hardware
//
#include <hip/hip_runtime.h>

typedef unsigned short u16;
typedef unsigned int   u32;
typedef __attribute__((ext_vector_type(16))) __bf16 v16bf;
typedef __attribute__((ext_vector_type(8)))  float  v8f;
typedef __attribute__((ext_vector_type(4)))  u32    u32x4;
typedef __attribute__((ext_vector_type(4)))  int    i32x4;

#define K_NN 20

#if defined(__AMDGCN__) && __has_builtin(__builtin_amdgcn_global_load_async_to_lds_b128)
#define HAVE_ASYNC_LDS 1
typedef __attribute__((address_space(1))) i32x4 gi32x4;   // global AS int4
typedef __attribute__((address_space(3))) i32x4 li32x4;   // LDS AS int4
#else
#define HAVE_ASYNC_LDS 0
#endif

__device__ __forceinline__ u16 f2bf(float f) {
    u32 u = __float_as_uint(f);
    u32 r = (u + 0x7FFFu + ((u >> 16) & 1u)) >> 16;
    return (u16)r;
}
__device__ __forceinline__ float gelu_exact(float x) {
    return 0.5f * x * (1.0f + erff(x * 0.70710678118654752f));
}

// ---------------------------------------------------------------------------
// WMMA bf16 GEMM:  out[M,N] = act( A[M,K] @ B[N,K]^T + bias[N] )
// A,B row-major bf16 (K contiguous). Block tile 128x128x32, 8 waves,
// each wave 32x64 (2x4 tiles of v_wmma_f32_16x16x32_bf16).
// Tiles staged in LDS via GLOBAL_LOAD_ASYNC_TO_LDS_B128 (ASYNCcnt) when
// available; fallback is global_load_b128 + ds_store_b128.
// ---------------------------------------------------------------------------
#define TBM 128
#define TBN 128
#define TBK 32
#define LDSK 48   // padded LDS row stride in halves (96B, 16B-multiple)

__global__ __launch_bounds__(256) void gemm_bf16_kernel(
    const u16* __restrict__ A, const u16* __restrict__ B,
    const float* __restrict__ bias,
    float* __restrict__ outF, u16* __restrict__ outH,
    int M, int N, int K, int act)
{
    __shared__ u16 As[TBM * LDSK];
    __shared__ u16 Bs[TBN * LDSK];

    const int tid  = threadIdx.x;
    const int m0   = blockIdx.y * TBM;
    const int n0   = blockIdx.x * TBN;
    const int wave = tid >> 5;
    const int lane = tid & 31;
    const int wM   = (wave & 3) * 32;   // 4 waves down
    const int wN   = (wave >> 2) * 64;  // 2 waves across
    const int fr   = lane & 15;         // row within 16x16 fragment
    const int kh   = (lane >> 4) * 8;   // K half-offset per ISA layout

    v8f acc[2][4] = {};

    union Frag { u32x4 q[2]; v16bf v; };

    for (int k0 = 0; k0 < K; k0 += TBK) {
        // cooperative tile loads: 128 rows x 32 halves = 512 16B chunks, 2/thread
        #pragma unroll
        for (int i = 0; i < 2; ++i) {
            int c = tid + i * 256;
            int r = c >> 2, cc = (c & 3) * 8;
#if HAVE_ASYNC_LDS
            __builtin_amdgcn_global_load_async_to_lds_b128(
                (gi32x4*)(A + (size_t)(m0 + r) * K + k0 + cc),
                (li32x4*)(As + r * LDSK + cc), 0, 0);
            __builtin_amdgcn_global_load_async_to_lds_b128(
                (gi32x4*)(B + (size_t)(n0 + r) * K + k0 + cc),
                (li32x4*)(Bs + r * LDSK + cc), 0, 0);
#else
            *(u32x4*)(As + r * LDSK + cc) =
                *(const u32x4*)(A + (size_t)(m0 + r) * K + k0 + cc);
            *(u32x4*)(Bs + r * LDSK + cc) =
                *(const u32x4*)(B + (size_t)(n0 + r) * K + k0 + cc);
#endif
        }
#if HAVE_ASYNC_LDS
#if __has_builtin(__builtin_amdgcn_s_wait_asynccnt)
        __builtin_amdgcn_s_wait_asynccnt(0);
#else
        asm volatile("s_wait_asynccnt 0x0" ::: "memory");
#endif
#endif
        __syncthreads();

        Frag aF[2], bF[4];
        #pragma unroll
        for (int mt = 0; mt < 2; ++mt) {
            const u16* p = As + (wM + mt * 16 + fr) * LDSK + kh;
            aF[mt].q[0] = *(const u32x4*)(p);
            aF[mt].q[1] = *(const u32x4*)(p + 16);
        }
        #pragma unroll
        for (int nt = 0; nt < 4; ++nt) {
            const u16* p = Bs + (wN + nt * 16 + fr) * LDSK + kh;
            bF[nt].q[0] = *(const u32x4*)(p);
            bF[nt].q[1] = *(const u32x4*)(p + 16);
        }
        #pragma unroll
        for (int mt = 0; mt < 2; ++mt)
            #pragma unroll
            for (int nt = 0; nt < 4; ++nt)
                acc[mt][nt] = __builtin_amdgcn_wmma_f32_16x16x32_bf16(
                    false, aF[mt].v, false, bF[nt].v,
                    (short)0, acc[mt][nt], false, false);
        __syncthreads();
    }

    // epilogue: C/D layout — VGPR r -> M = r + 8*(lane/16), N = lane%16
    const int rhi = (lane >> 4) * 8;
    #pragma unroll
    for (int mt = 0; mt < 2; ++mt) {
        #pragma unroll
        for (int nt = 0; nt < 4; ++nt) {
            #pragma unroll
            for (int r = 0; r < 8; ++r) {
                int m = m0 + wM + mt * 16 + rhi + r;
                int n = n0 + wN + nt * 16 + fr;
                float v = acc[mt][nt][r];
                if (bias) v += bias[n];
                if (act)  v = gelu_exact(v);
                if (outF) outF[(size_t)m * N + n] = v;
                else      outH[(size_t)m * N + n] = f2bf(v);
            }
        }
    }
}

// ---------------------------------------------------------------------------
// tnet per-point 3-layer MLP (3->64->128->1024), BN(eval)+ReLU, writes [1024,2048]
// ---------------------------------------------------------------------------
__global__ __launch_bounds__(256) void tnet_kernel(
    const float* __restrict__ cloud,
    const float* __restrict__ w0, const float* __restrict__ b0,
    const float* __restrict__ g0, const float* __restrict__ e0,
    const float* __restrict__ w1, const float* __restrict__ b1,
    const float* __restrict__ g1, const float* __restrict__ e1,
    const float* __restrict__ w2, const float* __restrict__ b2,
    const float* __restrict__ g2, const float* __restrict__ e2,
    float* __restrict__ feat)
{
    int n = blockIdx.x * blockDim.x + threadIdx.x;
    if (n >= 2048) return;
    const float RS = rsqrtf(1.0f + 1e-5f);
    float x0 = cloud[n], x1 = cloud[2048 + n], x2 = cloud[4096 + n];
    float h0[64];
    for (int o = 0; o < 64; ++o) {
        float s = w0[o * 3] * x0 + w0[o * 3 + 1] * x1 + w0[o * 3 + 2] * x2 + b0[o];
        s = s * (g0[o] * RS) + e0[o];
        h0[o] = fmaxf(s, 0.0f);
    }
    float h1[128];
    for (int o = 0; o < 128; ++o) {
        float s = b1[o];
        const float* w = w1 + o * 64;
        for (int c = 0; c < 64; ++c) s += w[c] * h0[c];
        s = s * (g1[o] * RS) + e1[o];
        h1[o] = fmaxf(s, 0.0f);
    }
    for (int o = 0; o < 1024; ++o) {
        float s = b2[o];
        const float* w = w2 + o * 128;
        for (int c = 0; c < 128; ++c) s += w[c] * h1[c];
        s = s * (g2[o] * RS) + e2[o];
        feat[(size_t)o * 2048 + n] = fmaxf(s, 0.0f);
    }
}

__global__ void maxreduce_kernel(const float* __restrict__ feat, float* __restrict__ hcat)
{
    int g = blockIdx.x * blockDim.x + threadIdx.x;
    if (g >= 4096) return;
    int bc = g >> 10, o = g & 1023;
    int b = bc >> 1, cl = bc & 1;
    const float* f = feat + ((size_t)bc * 1024 + o) * 2048;
    float m = -3.4e38f;
    for (int n = 0; n < 2048; ++n) m = fmaxf(m, f[n]);
    hcat[b * 2048 + cl * 1024 + o] = m;
}

__global__ __launch_bounds__(256) void head_kernel(
    const float* __restrict__ hcat,
    const float* __restrict__ w0, const float* __restrict__ b0,
    const float* __restrict__ w1, const float* __restrict__ b1,
    const float* __restrict__ w2, const float* __restrict__ b2,
    float* __restrict__ pose)
{
    __shared__ float s0[512];
    __shared__ float s1[256];
    int b = blockIdx.x, t = threadIdx.x;
    const float* h = hcat + b * 2048;
    for (int o = t; o < 512; o += 256) {
        float s = b0[o];
        const float* w = w0 + (size_t)o * 2048;
        for (int c = 0; c < 2048; ++c) s += w[c] * h[c];
        s0[o] = fmaxf(s, 0.0f);
    }
    __syncthreads();
    for (int o = t; o < 256; o += 256) {
        float s = b1[o];
        const float* w = w1 + (size_t)o * 512;
        for (int c = 0; c < 512; ++c) s += w[c] * s0[c];
        s1[o] = fmaxf(s, 0.0f);
    }
    __syncthreads();
    if (t < 6) {
        float s = b2[t];
        const float* w = w2 + (size_t)t * 256;
        for (int c = 0; c < 256; ++c) s += w[c] * s1[c];
        pose[b * 6 + t] = s;
    }
}

// Euler XYZ (Rx@Ry@Rz) rigid transform of cloud_a
__global__ void transform_kernel(const float* __restrict__ ca_in,
                                 const float* __restrict__ pose,
                                 float* __restrict__ at)
{
    int g = blockIdx.x * blockDim.x + threadIdx.x;
    if (g >= 4096) return;
    int b = g >> 11, n = g & 2047;
    const float* p = pose + b * 6;
    float ca = cosf(p[0]), sa = sinf(p[0]);
    float cb = cosf(p[1]), sb = sinf(p[1]);
    float cg = cosf(p[2]), sg = sinf(p[2]);
    float R00 = cb * cg,                R01 = -cb * sg,               R02 = sb;
    float R10 = ca * sg + sa * sb * cg, R11 = ca * cg - sa * sb * sg, R12 = -sa * cb;
    float R20 = sa * sg - ca * sb * cg, R21 = sa * cg + ca * sb * sg, R22 = ca * cb;
    const float* c = ca_in + (size_t)b * 6144;
    float x = c[n], y = c[2048 + n], z = c[4096 + n];
    float* o = at + (size_t)b * 6144;
    o[n]        = R00 * x + R01 * y + R02 * z + p[3];
    o[2048 + n] = R10 * x + R11 * y + R12 * z + p[4];
    o[4096 + n] = R20 * x + R21 * y + R22 * z + p[5];
}

// per-row top-20 of neg squared distance (self included, stable ties)
__global__ void knn_kernel(const float* __restrict__ pts, int* __restrict__ idx)
{
    int i = blockIdx.x * blockDim.x + threadIdx.x;
    if (i >= 2048) return;
    float xi = pts[i], yi = pts[2048 + i], zi = pts[4096 + i];
    float bd[K_NN]; int bi[K_NN];
    for (int t = 0; t < K_NN; ++t) { bd[t] = -3.4e38f; bi[t] = 0; }
    for (int j = 0; j < 2048; ++j) {
        float dx = pts[j] - xi, dy = pts[2048 + j] - yi, dz = pts[4096 + j] - zi;
        float d = -(dx * dx + dy * dy + dz * dz);
        if (d > bd[K_NN - 1]) {
            int p = K_NN - 1;
            while (p > 0 && bd[p - 1] < d) { bd[p] = bd[p - 1]; bi[p] = bi[p - 1]; --p; }
            bd[p] = d; bi[p] = j;
        }
    }
    for (int t = 0; t < K_NN; ++t) idx[i * K_NN + t] = bi[t];
}

// graph feature [nbr-ctr, ctr] -> 6x6 conv -> BN(eval) -> LeakyReLU(0.2)
__global__ void gfeat_kernel(const float* __restrict__ pts, const int* __restrict__ idx,
                             const float* __restrict__ gw, const float* __restrict__ gb,
                             const float* __restrict__ bng, const float* __restrict__ bnb,
                             float* __restrict__ tokens)
{
    int g = blockIdx.x * blockDim.x + threadIdx.x;
    if (g >= 2048 * K_NN) return;
    int n = g / K_NN;
    int j = idx[g];
    float f[6];
    f[3] = pts[n]; f[4] = pts[2048 + n]; f[5] = pts[4096 + n];
    f[0] = pts[j] - f[3]; f[1] = pts[2048 + j] - f[4]; f[2] = pts[4096 + j] - f[5];
    const float RS = rsqrtf(1.0f + 1e-5f);
    for (int o = 0; o < 6; ++o) {
        float s = gb[o];
        for (int c = 0; c < 6; ++c) s += gw[o * 6 + c] * f[c];
        s = s * (bng[o] * RS) + bnb[o];
        tokens[(size_t)g * 6 + o] = s > 0.0f ? s : 0.2f * s;
    }
}

// LayerNorm over C, write bf16 with optional zero-padded stride Cs
__global__ void ln_kernel(const float* __restrict__ x,
                          const float* __restrict__ g, const float* __restrict__ b,
                          u16* __restrict__ out, int M, int C, int Cs)
{
    int r = blockIdx.x * blockDim.x + threadIdx.x;
    if (r >= M) return;
    const float* xr = x + (size_t)r * C;
    float s = 0.0f, q = 0.0f;
    for (int c = 0; c < C; ++c) { float v = xr[c]; s += v; q += v * v; }
    float m = s / C;
    float var = q / C - m * m;
    float inv = rsqrtf(var + 1e-5f);
    u16* o = out + (size_t)r * Cs;
    for (int c = 0; c < C; ++c) o[c] = f2bf((xr[c] - m) * inv * g[c] + b[c]);
    for (int c = C; c < Cs; ++c) o[c] = 0;
}

// f32 [O,C] -> bf16 [O,Cs] zero-padded
__global__ void wconv_kernel(const float* __restrict__ src, u16* __restrict__ dst,
                             int O, int C, int Cs)
{
    int g = blockIdx.x * blockDim.x + threadIdx.x;
    if (g >= O * Cs) return;
    int o = g / Cs, c = g % Cs;
    dst[g] = (c < C) ? f2bf(src[(size_t)o * C + c]) : (u16)0;
}

// ---------------------------------------------------------------------------
extern "C" void kernel_launch(void* const* d_in, const int* in_sizes, int n_in,
                              void* d_out, int out_size, void* d_ws, size_t ws_size,
                              hipStream_t stream)
{
    (void)in_sizes; (void)n_in; (void)out_size; (void)ws_size;
    auto in = [&](int i) { return (const float*)d_in[i]; };

    const size_t M   = 40960;             // N*k tokens per (batch, branch)
    char* ws = (char*)d_ws;
    size_t off = 0;
    auto alloc = [&](size_t bytes) { size_t o = off; off = (off + bytes + 255) & ~(size_t)255; return o; };

    float* featT  = (float*)(ws + alloc(4ull * 1024 * 2048 * 4));
    float* hcat   = (float*)(ws + alloc(2 * 2048 * 4));
    float* pose   = (float*)(ws + alloc(2 * 6 * 4));
    float* a_t    = (float*)(ws + alloc(2 * 3 * 2048 * 4));
    int*   knn    = (int*)  (ws + alloc(2048ull * K_NN * 4));
    float* tokens = (float*)(ws + alloc(M * 6 * 4));
    u16*   xln    = (u16*)  (ws + alloc(M * 32 * 2));
    u16*   BIGH   = (u16*)  (ws + alloc(M * 1024 * 2));
    float* BIGF   = (float*)(ws + alloc(M * 1024 * 4));
    u16*   S1     = (u16*)  (ws + alloc(M * 128 * 2));
    float* S2     = (float*)(ws + alloc(M * 128 * 4));
    u16* w_m1t1 = (u16*)(ws + alloc(1024 * 32 * 2));
    u16* w_m1t2 = (u16*)(ws + alloc(1024ull * 1024 * 2));
    u16* w_m1c1 = (u16*)(ws + alloc(128ull * 1024 * 2));
    u16* w_m1c2 = (u16*)(ws + alloc(128 * 128 * 2));
    u16* w_m2t1 = (u16*)(ws + alloc(1024 * 128 * 2));
    u16* w_m2t2 = (u16*)(ws + alloc(1024ull * 1024 * 2));
    u16* w_m2c1 = (u16*)(ws + alloc(128ull * 1024 * 2));
    u16* w_m2c2 = (u16*)(ws + alloc(128 * 128 * 2));

    const float* clouds[2] = { (const float*)d_in[0], (const float*)d_in[1] };

    // ---- tnet features + global max pooling -> hcat[2,2048] ----
    for (int b = 0; b < 2; ++b)
        for (int cl = 0; cl < 2; ++cl)
            tnet_kernel<<<8, 256, 0, stream>>>(
                clouds[cl] + (size_t)b * 6144,
                in(2), in(3), in(4), in(5), in(6), in(7), in(8), in(9),
                in(10), in(11), in(12), in(13),
                featT + (size_t)(b * 2 + cl) * 1024 * 2048);
    maxreduce_kernel<<<16, 256, 0, stream>>>(featT, hcat);

    // ---- head MLP -> pose; rigid transform of cloud_a ----
    head_kernel<<<2, 256, 0, stream>>>(hcat, in(14), in(15), in(16), in(17), in(18), in(19), pose);
    transform_kernel<<<16, 256, 0, stream>>>((const float*)d_in[0], pose, a_t);

    // ---- weight conversion to bf16 (K-padded where needed) ----
    auto wc = [&](int src, u16* dst, int O, int C, int Cs) {
        int tot = O * Cs;
        wconv_kernel<<<(tot + 255) / 256, 256, 0, stream>>>(in(src), dst, O, C, Cs);
    };
    wc(26, w_m1t1, 1024, 6, 32);
    wc(28, w_m1t2, 1024, 1024, 1024);
    wc(32, w_m1c1, 128, 1024, 1024);
    wc(34, w_m1c2, 128, 128, 128);
    wc(38, w_m2t1, 1024, 128, 128);
    wc(40, w_m2t2, 1024, 1024, 1024);
    wc(44, w_m2c1, 128, 1024, 1024);
    wc(46, w_m2c2, 128, 128, 128);

    dim3 g8(1024 / TBN, M / TBM);   // N=1024 GEMMs
    dim3 g1(128 / TBN,  M / TBM);   // N=128  GEMMs
    const int LNB = (int)((M + 255) / 256);

    // ---- per (branch, batch) pipeline ----
    for (int br = 0; br < 2; ++br) {
        for (int b = 0; b < 2; ++b) {
            const float* pts = (br == 0) ? (a_t + (size_t)b * 6144)
                                         : ((const float*)d_in[1] + (size_t)b * 6144);
            knn_kernel<<<8, 256, 0, stream>>>(pts, knn);
            gfeat_kernel<<<160, 256, 0, stream>>>(pts, knn, in(20), in(21), in(22), in(23), tokens);

            // mixer m1
            ln_kernel<<<LNB, 256, 0, stream>>>(tokens, in(24), in(25), xln, (int)M, 6, 32);
            gemm_bf16_kernel<<<g8, 256, 0, stream>>>(xln, w_m1t1, in(27), nullptr, BIGH, (int)M, 1024, 32, 1);
            gemm_bf16_kernel<<<g8, 256, 0, stream>>>(BIGH, w_m1t2, in(29), BIGF, nullptr, (int)M, 1024, 1024, 0);
            ln_kernel<<<LNB, 256, 0, stream>>>(BIGF, in(30), in(31), BIGH, (int)M, 1024, 1024);
            gemm_bf16_kernel<<<g1, 256, 0, stream>>>(BIGH, w_m1c1, in(33), nullptr, S1, (int)M, 128, 1024, 1);
            gemm_bf16_kernel<<<g1, 256, 0, stream>>>(S1, w_m1c2, in(35), S2, nullptr, (int)M, 128, 128, 0);

            // mixer m2
            ln_kernel<<<LNB, 256, 0, stream>>>(S2, in(36), in(37), S1, (int)M, 128, 128);
            gemm_bf16_kernel<<<g8, 256, 0, stream>>>(S1, w_m2t1, in(39), nullptr, BIGH, (int)M, 1024, 128, 1);
            gemm_bf16_kernel<<<g8, 256, 0, stream>>>(BIGH, w_m2t2, in(41), BIGF, nullptr, (int)M, 1024, 1024, 0);
            ln_kernel<<<LNB, 256, 0, stream>>>(BIGF, in(42), in(43), BIGH, (int)M, 1024, 1024);
            gemm_bf16_kernel<<<g1, 256, 0, stream>>>(BIGH, w_m2c1, in(45), nullptr, S1, (int)M, 128, 1024, 1);

            float* outp = (float*)d_out + ((size_t)br * 2 + b) * M * 128;
            gemm_bf16_kernel<<<g1, 256, 0, stream>>>(S1, w_m2c2, in(47), outp, nullptr, (int)M, 128, 128, 0);
        }
    }
}
